// Video_Counter_35837207118364
// MI455X (gfx1250) — matrix operand order repaired and checked by reading, not compiled, not run on hardware
//
#include <hip/hip_runtime.h>
#include <hip/hip_bf16.h>

// ---------------------------------------------------------------------------
// Transformer block (LN -> QKV -> masked attention -> proj -> LN -> MLP) for
// MI455X / gfx1250.  All GEMM-like math uses v_wmma_f32_16x16x32_f16 (f16 in,
// f32 accumulate).  Wave32 layouts follow CDNA5 ISA 7.12.2:
//   A (16x32 f16): lane m=lane&15; halfs 0..7 -> K = hi*8+0..7,
//                  halfs 8..15 -> K = 16+hi*8+0..7   (hi = lane>=16)
//   B (32x16 f16): lane n=lane&15; same K mapping as A
//   C/D (16x16 f32): elem j -> row = j + hi*8, col = lane&15
// GEMMs are 2x2 register-blocked: each wave owns a 32x32 output tile
// (4 accumulators), so 4 WMMAs share 8 b128 loads instead of 4 sharing 16.
// ---------------------------------------------------------------------------

typedef __attribute__((ext_vector_type(16))) _Float16 v16h;
typedef __attribute__((ext_vector_type(8)))  _Float16 v8h;
typedef __attribute__((ext_vector_type(8)))  float    v8f;

#define SEQ_N   2304
#define DIM_C   256
#define NHEADS  8
#define HEADD   32
#define MLPH    1024
#define BATCH   4
#define ROWS    (BATCH * SEQ_N)      // 9216
#define GRID_W  48
#define ATT_SCALE 0.17677669529663687f   // 32^-0.5
#define NEGBIG  -1e9f

// ---- fragment loaders ------------------------------------------------------

__device__ inline v16h load_a16(const _Float16* base, int ld, int lane) {
  int m = lane & 15, hi = lane >> 4;
  const _Float16* p = base + (size_t)m * ld + hi * 8;
  v8h lo = *(const v8h*)(p);
  v8h hh = *(const v8h*)(p + 16);
  v16h r;
#pragma unroll
  for (int i = 0; i < 8; ++i) { r[i] = lo[i]; r[i + 8] = hh[i]; }
  return r;
}

__device__ inline v16h load_b16(const _Float16* base, int ld, int lane) {
  int n = lane & 15, hi = lane >> 4;
  const _Float16* p = base + (size_t)n * ld + hi * 8;
  v8h lo = *(const v8h*)(p);
  v8h hh = *(const v8h*)(p + 16);
  v16h r;
#pragma unroll
  for (int i = 0; i < 8; ++i) { r[i] = lo[i]; r[i + 8] = hh[i]; }
  return r;
}

__device__ inline float rmax16(float v) {
#pragma unroll
  for (int m = 1; m < 16; m <<= 1) v = fmaxf(v, __shfl_xor(v, m, 32));
  return v;
}
__device__ inline float rsum16(float v) {
#pragma unroll
  for (int m = 1; m < 16; m <<= 1) v += __shfl_xor(v, m, 32);
  return v;
}

// ---- weight transpose + f32->f16 convert:  Wt[n*K+k] = W[k*N+n] ------------

__global__ void __launch_bounds__(256)
wt_transpose_kernel(const float* __restrict__ W, _Float16* __restrict__ Wt,
                    int K, int Nn) {
  int i = blockIdx.x * 256 + threadIdx.x;
  if (i >= K * Nn) return;
  int n = i / K, k = i - n * K;
  Wt[i] = (_Float16)W[(size_t)k * Nn + n];
}

// ---- LayerNorm (one row of C=256 per 256-thread block), f16 out ------------

__global__ void __launch_bounds__(256)
ln_f16_kernel(const float* __restrict__ x, const float* __restrict__ g,
              const float* __restrict__ b, _Float16* __restrict__ out) {
  __shared__ float sred[256];
  int row = blockIdx.x, tid = threadIdx.x;
  float v = x[(size_t)row * DIM_C + tid];
  sred[tid] = v;
  __syncthreads();
#pragma unroll
  for (int s = 128; s > 0; s >>= 1) {
    if (tid < s) sred[tid] += sred[tid + s];
    __syncthreads();
  }
  float mu = sred[0] * (1.0f / DIM_C);
  __syncthreads();
  float d = v - mu;
  sred[tid] = d * d;
  __syncthreads();
#pragma unroll
  for (int s = 128; s > 0; s >>= 1) {
    if (tid < s) sred[tid] += sred[tid + s];
    __syncthreads();
  }
  float var = sred[0] * (1.0f / DIM_C);
  float y = d * rsqrtf(var + 1e-6f) * g[tid] + b[tid];
  out[(size_t)row * DIM_C + tid] = (_Float16)y;
}

// ---- epilogue for one 16x16 D tile -----------------------------------------
// mode 0: store f16            mode 1: GELU then store f16
// mode 2: + resid, store f32   mode 3: store f16 scattered as V-transpose

__device__ inline void store_tile(const v8f& acc, int r0, int c0, int lane,
                                  const float* __restrict__ bias,
                                  const float* __restrict__ resid,
                                  float* __restrict__ out32,
                                  _Float16* __restrict__ out16,
                                  int Nn, int mode) {
  int n = lane & 15, hi = lane >> 4;
#pragma unroll
  for (int j = 0; j < 8; ++j) {
    int r = r0 + j + hi * 8;
    int c = c0 + n;
    float v = acc[j] + bias[c];
    if (mode == 2) {
      size_t idx = (size_t)r * Nn + c;
      out32[idx] = v + resid[idx];
    } else {
      if (mode == 1) v = 0.5f * v * (1.0f + erff(v * 0.70710678118654752f));
      if (mode == 3) {
        int bi = r / SEQ_N, ni = r - bi * SEQ_N;
        int hh = c >> 5, d = c & 31;
        out16[(((size_t)bi * NHEADS + hh) * HEADD + d) * SEQ_N + ni] = (_Float16)v;
      } else {
        out16[(size_t)r * Nn + c] = (_Float16)v;
      }
    }
  }
}

// ---- 2x2 register-blocked WMMA GEMM ----------------------------------------
// out = A[f16, MxK] * Bt[f16, NxK]^T + bias, with epilogue per `mode`.
// Block = 256 threads = 8 waves arranged 2 (rows) x 4 (cols):
//   block tile 64 x 128, wave tile 32 x 32.

__global__ void __launch_bounds__(256)
gemm_wmma_kernel(const _Float16* __restrict__ A, const _Float16* __restrict__ Bt,
                 const float* __restrict__ bias, const float* __restrict__ resid,
                 float* __restrict__ out32, _Float16* __restrict__ out16,
                 int M, int Nn, int K, int mode) {
  int lane = threadIdx.x & 31;
  int w = threadIdx.x >> 5;
  int row0 = blockIdx.x * 64 + (w & 1) * 32;
  int col0 = blockIdx.y * 128 + (w >> 1) * 32;
  if (row0 >= M || col0 >= Nn) return;

  const _Float16* ap = A + (size_t)row0 * K;
  const _Float16* bp = Bt + (size_t)col0 * K;
  v8f acc00 = {}, acc01 = {}, acc10 = {}, acc11 = {};
  for (int k = 0; k < K; k += 32) {
    __builtin_prefetch(ap + k + 256, 0, 1);   // global_prefetch next A chunk
    __builtin_prefetch(bp + k + 256, 0, 1);   // global_prefetch next B chunk
    v16h a0 = load_a16(ap + k, K, lane);
    v16h a1 = load_a16(ap + (size_t)16 * K + k, K, lane);
    v16h b0 = load_b16(bp + k, K, lane);
    v16h b1 = load_b16(bp + (size_t)16 * K + k, K, lane);
    acc00 = __builtin_amdgcn_wmma_f32_16x16x32_f16(false, a0, false, b0,
                                                   (short)0, acc00, false, false);
    acc01 = __builtin_amdgcn_wmma_f32_16x16x32_f16(false, a0, false, b1,
                                                   (short)0, acc01, false, false);
    acc10 = __builtin_amdgcn_wmma_f32_16x16x32_f16(false, a1, false, b0,
                                                   (short)0, acc10, false, false);
    acc11 = __builtin_amdgcn_wmma_f32_16x16x32_f16(false, a1, false, b1,
                                                   (short)0, acc11, false, false);
  }

  store_tile(acc00, row0,      col0,      lane, bias, resid, out32, out16, Nn, mode);
  store_tile(acc01, row0,      col0 + 16, lane, bias, resid, out32, out16, Nn, mode);
  store_tile(acc10, row0 + 16, col0,      lane, bias, resid, out32, out16, Nn, mode);
  store_tile(acc11, row0 + 16, col0 + 16, lane, bias, resid, out32, out16, Nn, mode);
}

// ---- flash-style masked attention: one 16-row q-tile per wave --------------
// Q,K: f16 [B*SEQ_N, C] (head h at cols h*32..h*32+31).  Vt: f16 V-transpose.

__global__ void __launch_bounds__(256)
attn_wmma_kernel(const _Float16* __restrict__ Q, const _Float16* __restrict__ Kf,
                 const _Float16* __restrict__ Vt, _Float16* __restrict__ O) {
  __shared__ __align__(16) _Float16 ptile[8][16 * 32];
  int lane = threadIdx.x & 31;
  int w = threadIdx.x >> 5;
  int tile = blockIdx.x * 8 + w;
  const int NTQ = SEQ_N / 16;                 // 144 q-tiles per head
  int q0 = (tile % NTQ) * 16;
  int h  = (tile / NTQ) % NHEADS;
  int b  = tile / (NTQ * NHEADS);
  int n = lane & 15, hi = lane >> 4;
  _Float16* lds = &ptile[w][0];

  const _Float16* qbase = Q + ((size_t)(b * SEQ_N + q0)) * DIM_C + h * HEADD;
  v16h qa = load_a16(qbase, DIM_C, lane);

  int qy[8], qx[8];
#pragma unroll
  for (int j = 0; j < 8; ++j) {
    int qi = q0 + j + hi * 8;
    qy[j] = qi / GRID_W; qx[j] = qi - qy[j] * GRID_W;
  }

  float mrun[8], lrun[8];
  v8f o0 = {}, o1 = {};
#pragma unroll
  for (int j = 0; j < 8; ++j) { mrun[j] = -1e30f; lrun[j] = 0.0f; }

  const _Float16* vbase =
      Vt + ((size_t)(b * NHEADS + h)) * HEADD * SEQ_N;   // [HD][SEQ_N]

  for (int kt = 0; kt < SEQ_N; kt += 32) {
    const _Float16* kb = Kf + ((size_t)(b * SEQ_N + kt)) * DIM_C + h * HEADD;
    v16h kb0 = load_b16(kb, DIM_C, lane);
    v16h kb1 = load_b16(kb + 16 * DIM_C, DIM_C, lane);
    v8f z = {};
    v8f s0 = __builtin_amdgcn_wmma_f32_16x16x32_f16(false, qa, false, kb0,
                                                    (short)0, z, false, false);
    v8f s1 = __builtin_amdgcn_wmma_f32_16x16x32_f16(false, qa, false, kb1,
                                                    (short)0, z, false, false);
    int k0 = kt + n, k1 = kt + 16 + n;
    int ky0 = k0 / GRID_W, kx0 = k0 - ky0 * GRID_W;
    int ky1 = k1 / GRID_W, kx1 = k1 - ky1 * GRID_W;

#pragma unroll
    for (int j = 0; j < 8; ++j) {
      int dy0 = qy[j] - ky0, dx0 = qx[j] - kx0;
      int dy1 = qy[j] - ky1, dx1 = qx[j] - kx1;
      float sv0 = s0[j] * ATT_SCALE + ((dy0 * dy0 + dx0 * dx0 > 9) ? NEGBIG : 0.0f);
      float sv1 = s1[j] * ATT_SCALE + ((dy1 * dy1 + dx1 * dx1 > 9) ? NEGBIG : 0.0f);
      float t = rmax16(fmaxf(sv0, sv1));
      float nm = fmaxf(mrun[j], t);
      float f = __expf(mrun[j] - nm);
      mrun[j] = nm;
      float p0 = __expf(sv0 - nm), p1 = __expf(sv1 - nm);
      lrun[j] = lrun[j] * f + rsum16(p0 + p1);
      o0[j] *= f; o1[j] *= f;
      int rr = j + hi * 8;                        // D-layout row
      lds[rr * 32 + n]      = (_Float16)p0;       // re-shape D -> A via LDS
      lds[rr * 32 + 16 + n] = (_Float16)p1;
    }
    asm volatile("s_wait_dscnt 0x0" ::: "memory"); // stores visible to loads

    v16h pa = load_a16(lds, 32, lane);            // P as A-fragment [16x32]
    const _Float16* vb = vbase + kt;
    v16h vb0 = load_b16(vb, SEQ_N, lane);                 // dims 0..15
    v16h vb1 = load_b16(vb + (size_t)16 * SEQ_N, SEQ_N, lane); // dims 16..31
    o0 = __builtin_amdgcn_wmma_f32_16x16x32_f16(false, pa, false, vb0,
                                                (short)0, o0, false, false);
    o1 = __builtin_amdgcn_wmma_f32_16x16x32_f16(false, pa, false, vb1,
                                                (short)0, o1, false, false);
  }

#pragma unroll
  for (int j = 0; j < 8; ++j) {
    int r = q0 + j + hi * 8;
    size_t rowoff = ((size_t)(b * SEQ_N + r)) * DIM_C + h * HEADD;
    float inv = 1.0f / lrun[j];
    O[rowoff + n]      = (_Float16)(o0[j] * inv);
    O[rowoff + 16 + n] = (_Float16)(o1[j] * inv);
  }
}

// ---------------------------------------------------------------------------

extern "C" void kernel_launch(void* const* d_in, const int* in_sizes, int n_in,
                              void* d_out, int out_size, void* d_ws, size_t ws_size,
                              hipStream_t stream) {
  const float* x    = (const float*)d_in[0];
  const float* x_kv = (const float*)d_in[1];
  const float* Wq   = (const float*)d_in[2];
  const float* bq   = (const float*)d_in[3];
  const float* Wk   = (const float*)d_in[4];
  const float* bk   = (const float*)d_in[5];
  const float* Wv   = (const float*)d_in[6];
  const float* bv   = (const float*)d_in[7];
  const float* Wp   = (const float*)d_in[8];
  const float* bp   = (const float*)d_in[9];
  const float* g1   = (const float*)d_in[10];
  const float* b1   = (const float*)d_in[11];
  const float* g2   = (const float*)d_in[12];
  const float* b2   = (const float*)d_in[13];
  const float* W1   = (const float*)d_in[14];
  const float* bm1  = (const float*)d_in[15];
  const float* W2   = (const float*)d_in[16];
  const float* bm2  = (const float*)d_in[17];
  float* out = (float*)d_out;

  char* wsb = (char*)d_ws;
  size_t off = 0;
  auto alloc = [&](size_t bytes) -> char* {
    char* p = wsb + off;
    off += (bytes + 255) & ~(size_t)255;
    return p;
  };
  _Float16* WqT  = (_Float16*)alloc((size_t)DIM_C * DIM_C * 2);
  _Float16* WkT  = (_Float16*)alloc((size_t)DIM_C * DIM_C * 2);
  _Float16* WvT  = (_Float16*)alloc((size_t)DIM_C * DIM_C * 2);
  _Float16* WpT  = (_Float16*)alloc((size_t)DIM_C * DIM_C * 2);
  _Float16* W1T  = (_Float16*)alloc((size_t)DIM_C * MLPH * 2);
  _Float16* W2T  = (_Float16*)alloc((size_t)MLPH * DIM_C * 2);
  _Float16* h16  = (_Float16*)alloc((size_t)ROWS * DIM_C * 2);
  _Float16* hk16 = (_Float16*)alloc((size_t)ROWS * DIM_C * 2);
  _Float16* q16  = (_Float16*)alloc((size_t)ROWS * DIM_C * 2);
  _Float16* k16  = (_Float16*)alloc((size_t)ROWS * DIM_C * 2);
  _Float16* vt16 = (_Float16*)alloc((size_t)ROWS * DIM_C * 2);
  _Float16* att16= (_Float16*)alloc((size_t)ROWS * DIM_C * 2);
  _Float16* m16  = (_Float16*)alloc((size_t)ROWS * DIM_C * 2);
  _Float16* hm16 = (_Float16*)alloc((size_t)ROWS * MLPH * 2);
  float*    x1   = (float*)   alloc((size_t)ROWS * DIM_C * 4);

  // 1. weight transpose + f16 convert
  wt_transpose_kernel<<<(DIM_C * DIM_C + 255) / 256, 256, 0, stream>>>(Wq, WqT, DIM_C, DIM_C);
  wt_transpose_kernel<<<(DIM_C * DIM_C + 255) / 256, 256, 0, stream>>>(Wk, WkT, DIM_C, DIM_C);
  wt_transpose_kernel<<<(DIM_C * DIM_C + 255) / 256, 256, 0, stream>>>(Wv, WvT, DIM_C, DIM_C);
  wt_transpose_kernel<<<(DIM_C * DIM_C + 255) / 256, 256, 0, stream>>>(Wp, WpT, DIM_C, DIM_C);
  wt_transpose_kernel<<<(DIM_C * MLPH + 255) / 256, 256, 0, stream>>>(W1, W1T, DIM_C, MLPH);
  wt_transpose_kernel<<<(MLPH * DIM_C + 255) / 256, 256, 0, stream>>>(W2, W2T, MLPH, DIM_C);

  // 2. LayerNorm of both streams -> f16
  ln_f16_kernel<<<ROWS, 256, 0, stream>>>(x, g1, b1, h16);
  ln_f16_kernel<<<ROWS, 256, 0, stream>>>(x_kv, g1, b1, hk16);

  // 3. Q, K, V projections (WMMA)
  dim3 gC(ROWS / 64, DIM_C / 128);
  gemm_wmma_kernel<<<gC, 256, 0, stream>>>(h16, WqT, bq, nullptr, nullptr, q16,
                                           ROWS, DIM_C, DIM_C, 0);
  gemm_wmma_kernel<<<gC, 256, 0, stream>>>(hk16, WkT, bk, nullptr, nullptr, k16,
                                           ROWS, DIM_C, DIM_C, 0);
  gemm_wmma_kernel<<<gC, 256, 0, stream>>>(hk16, WvT, bv, nullptr, nullptr, vt16,
                                           ROWS, DIM_C, DIM_C, 3);

  // 4. masked flash attention (WMMA)
  attn_wmma_kernel<<<(BATCH * NHEADS * (SEQ_N / 16)) / 8, 256, 0, stream>>>(
      q16, k16, vt16, att16);

  // 5. output projection + residual (fp32)
  gemm_wmma_kernel<<<gC, 256, 0, stream>>>(att16, WpT, bp, x, x1, nullptr,
                                           ROWS, DIM_C, DIM_C, 2);

  // 6. LN2 -> f16
  ln_f16_kernel<<<ROWS, 256, 0, stream>>>(x1, g2, b2, m16);

  // 7. MLP up-projection + exact GELU
  dim3 gH(ROWS / 64, MLPH / 128);
  gemm_wmma_kernel<<<gH, 256, 0, stream>>>(m16, W1T, bm1, nullptr, nullptr, hm16,
                                           ROWS, MLPH, DIM_C, 1);

  // 8. MLP down-projection + residual -> d_out (fp32)
  gemm_wmma_kernel<<<gC, 256, 0, stream>>>(hm16, W2T, bm2, x1, out, nullptr,
                                           ROWS, DIM_C, MLPH, 2);
}
